// MixtureOfExperts_83597243449344
// MI455X (gfx1250) — matrix operand order repaired and checked by reading, not compiled, not run on hardware
//
#include <hip/hip_runtime.h>
#include <hip/hip_bf16.h>

// ---------------------------------------------------------------------------
// MoE top-2 routed GEMM for gfx1250 (wave32, WMMA bf16 16x16x32, f32 accum)
// N=8192 tokens, E=8 experts, IN=2048, OUT=2048
// Pipeline: [zero] -> [gate/route] -> [cvt X,W -> bf16 in ws] -> [grouped GEMM]
// ---------------------------------------------------------------------------

typedef __attribute__((ext_vector_type(16))) __bf16 v16bf;
typedef __attribute__((ext_vector_type(8)))  float  v8f;

// exact pointer types for the async global->LDS builtin (from clang diagnostic)
typedef int v4i_vs __attribute__((vector_size(16)));
typedef __attribute__((address_space(1))) v4i_vs* gv4i_p;   // global ("__device__")
typedef __attribute__((address_space(3))) v4i_vs* lv4i_p;   // LDS ("__shared__")

#define MOE_N    8192
#define MOE_E    8
#define MOE_IN   2048
#define MOE_OUT  2048
#define BM       128   // tokens per block tile
#define BN       128   // outputs per block tile
#define KB       32    // K chunk (matches wmma 16x16x32 K depth)
#define KIT      (MOE_IN / KB)
#define LDSK     48    // padded LDS row stride in halfs (16B aligned, bank-skewed)
#define ROWQ     (LDSK / 8)   // uint4 per LDS row (6)

union FragU {
    uint4 q[2];
    v16bf v;
};

// round-to-nearest-even f32 -> bf16 pair packed into a uint
__device__ __forceinline__ unsigned pack_bf16x2(float a, float b) {
    unsigned ua = __float_as_uint(a);
    unsigned ub = __float_as_uint(b);
    ua += 0x7FFFu + ((ua >> 16) & 1u);
    ub += 0x7FFFu + ((ub >> 16) & 1u);
    return (ua >> 16) | (ub & 0xFFFF0000u);
}

__device__ __forceinline__ void wait_async0() {
#if __has_builtin(__builtin_amdgcn_s_wait_asynccnt)
    __builtin_amdgcn_s_wait_asynccnt(0);
#else
    asm volatile("s_wait_asynccnt 0x0" ::: "memory");
#endif
}

// ---------------------------------------------------------------------------
// Kernel 1: zero output + routing counters
// ---------------------------------------------------------------------------
__global__ void moe_zero(float4* __restrict__ Y4, unsigned n4, int* __restrict__ counts) {
    unsigned i = blockIdx.x * blockDim.x + threadIdx.x;
    if (i < n4) Y4[i] = make_float4(0.f, 0.f, 0.f, 0.f);
    if (blockIdx.x == 0 && threadIdx.x < 16) counts[threadIdx.x] = 0;
}

// ---------------------------------------------------------------------------
// Kernel 2: gating — top-2 + renorm (softmax denom cancels), scatter buckets
// ---------------------------------------------------------------------------
__global__ void moe_gate(const float* __restrict__ G,
                         int* __restrict__ counts,
                         int* __restrict__ tokIds,
                         float* __restrict__ tokWt) {
    int n = blockIdx.x * blockDim.x + threadIdx.x;
    if (n >= MOE_N) return;

    float g[MOE_E];
#pragma unroll
    for (int e = 0; e < MOE_E; ++e) g[e] = G[n * MOE_E + e];

    float v0 = -1e30f, v1 = -1e30f;
    int   i0 = 0,      i1 = 0;
#pragma unroll
    for (int e = 0; e < MOE_E; ++e) {
        float x = g[e];
        if (x > v0)      { v1 = v0; i1 = i0; v0 = x; i0 = e; }
        else if (x > v1) { v1 = x; i1 = e; }
    }
    float e1  = __expf(v1 - v0);
    float inv = 1.0f / (1.0f + e1);

    int p0 = atomicAdd(&counts[i0], 1);
    tokIds[i0 * MOE_N + p0] = n;
    tokWt [i0 * MOE_N + p0] = inv;
    int p1 = atomicAdd(&counts[i1], 1);
    tokIds[i1 * MOE_N + p1] = n;
    tokWt [i1 * MOE_N + p1] = e1 * inv;
}

// ---------------------------------------------------------------------------
// Kernel 2b: bulk f32 -> packed bf16 conversion (one-time, memory bound)
// ---------------------------------------------------------------------------
__global__ void moe_cvt(const float4* __restrict__ src, uint2* __restrict__ dst,
                        unsigned n4) {
    unsigned i = blockIdx.x * blockDim.x + threadIdx.x;
    if (i < n4) {
        float4 v = src[i];
        dst[i] = make_uint2(pack_bf16x2(v.x, v.y), pack_bf16x2(v.z, v.w));
    }
}

// ---------------------------------------------------------------------------
// Shared epilogue helper: D tile 16x16 f32 -> weighted atomic accumulation
// VGPR j -> M = j + 8*(lane>>4), N = lane&15
// ---------------------------------------------------------------------------
__device__ __forceinline__ void moe_epilogue(
    const v8f* acc, const int* sTok, const float* sWt,
    const float* __restrict__ bvec, float* __restrict__ Y,
    int e, int tBase, int nBase, int cnt, int wave, int l15, int half16) {
    int   tokr[8];
    float wtr[8];
    bool  valr[8];
#pragma unroll
    for (int j = 0; j < 8; ++j) {
        int r   = wave * 16 + j + 8 * half16;
        tokr[j] = sTok[r];
        wtr[j]  = sWt[r];
        valr[j] = (tBase + r) < cnt;
    }
    float biasv[8];
#pragma unroll
    for (int nsub = 0; nsub < 8; ++nsub)
        biasv[nsub] = bvec[e * MOE_OUT + nBase + nsub * 16 + l15];

#pragma unroll
    for (int j = 0; j < 8; ++j) {
        if (valr[j]) {
            float* yrow = Y + (unsigned long long)tokr[j] * MOE_OUT + nBase + l15;
            float  w    = wtr[j];
#pragma unroll
            for (int nsub = 0; nsub < 8; ++nsub)
                atomicAdd(yrow + nsub * 16, w * (acc[nsub][j] + biasv[nsub]));
        }
    }
}

// ---------------------------------------------------------------------------
// Kernel 3 (fast path): grouped GEMM over pre-converted bf16 X/W.
// grid = (OUT/BN, N/BM, E), block = 256 threads = 8 wave32.
// Double-buffered LDS fed by async global->LDS copies (ASYNCcnt) when the
// toolchain exposes the builtin, else vector load + ds_store_b128.
// ---------------------------------------------------------------------------
__global__ void __launch_bounds__(256, 2)
moe_gemm_bf16(const char* __restrict__ Xb,   // [N][IN] bf16
              const char* __restrict__ Wb,   // [E][OUT][IN] bf16
              const float* __restrict__ bvec,
              const int*   __restrict__ counts,
              const int*   __restrict__ tokIds,
              const float* __restrict__ tokWt,
              float*       __restrict__ Y) {
    __shared__ uint4 Xs[2][BM * ROWQ];
    __shared__ uint4 Ws[2][BN * ROWQ];
    __shared__ int   sTok[BM];
    __shared__ float sWt[BM];
    __shared__ int   sCnt;

    const int e     = blockIdx.z;
    const int tBase = blockIdx.y * BM;
    const int nBase = blockIdx.x * BN;

    if (threadIdx.x == 0) sCnt = counts[e];
    __syncthreads();
    const int cnt = sCnt;
    if (tBase >= cnt) return;

    if (threadIdx.x < BM) {
        int r = tBase + threadIdx.x;
        if (r < cnt) {
            sTok[threadIdx.x] = tokIds[e * MOE_N + r];
            sWt [threadIdx.x] = tokWt [e * MOE_N + r];
        } else {
            sTok[threadIdx.x] = 0;
            sWt [threadIdx.x] = 0.f;
        }
    }
    __syncthreads();

    const int lane   = threadIdx.x & 31;
    const int wave   = threadIdx.x >> 5;
    const int l15    = lane & 15;
    const int half16 = lane >> 4;

    // fill coords: tile = 128 rows x 64B = 512 x 16B chunks per matrix,
    // 2 chunks per thread per matrix. chunk c: row = c>>2, q = c&3.
    char*   xsrc[2];
    char*   wsrc[2];
    int     lidx[2];
#pragma unroll
    for (int i = 0; i < 2; ++i) {
        int c   = threadIdx.x + i * 256;
        int row = c >> 2;
        int q   = c & 3;
        xsrc[i] = const_cast<char*>(Xb) +
                  (unsigned long long)sTok[row] * (MOE_IN * 2) + q * 16;
        wsrc[i] = const_cast<char*>(Wb) +
                  ((unsigned long long)e * MOE_OUT + nBase + row) * (MOE_IN * 2) + q * 16;
        lidx[i] = row * ROWQ + q;
    }

#if __has_builtin(__builtin_amdgcn_global_load_async_to_lds_b128)
    auto fill = [&](int buf, int k0) {
        const int kb = k0 * 2;   // byte offset along K
#pragma unroll
        for (int i = 0; i < 2; ++i) {
            __builtin_amdgcn_global_load_async_to_lds_b128(
                (gv4i_p)(xsrc[i] + kb), (lv4i_p)&Xs[buf][lidx[i]], 0, 0);
            __builtin_amdgcn_global_load_async_to_lds_b128(
                (gv4i_p)(wsrc[i] + kb), (lv4i_p)&Ws[buf][lidx[i]], 0, 0);
        }
    };
#else
    auto fill = [&](int buf, int k0) {
        const int kb = k0 * 2;
#pragma unroll
        for (int i = 0; i < 2; ++i) {
            Xs[buf][lidx[i]] = *(const uint4*)(xsrc[i] + kb);
            Ws[buf][lidx[i]] = *(const uint4*)(wsrc[i] + kb);
        }
    };
#endif

    v8f acc[8];
    const v8f vzero = {0.f, 0.f, 0.f, 0.f, 0.f, 0.f, 0.f, 0.f};
#pragma unroll
    for (int t = 0; t < 8; ++t) acc[t] = vzero;

    const int arow = (wave * 16 + l15) * ROWQ;

    fill(0, 0);
#if __has_builtin(__builtin_amdgcn_global_load_async_to_lds_b128)
    wait_async0();
#endif
    __syncthreads();

#pragma unroll 1
    for (int kk = 0; kk < KIT; ++kk) {
        const int cur = kk & 1;
        // issue next tile's copies first so they overlap the WMMAs below
        if (kk + 1 < KIT) fill((kk + 1) & 1, (kk + 1) * KB);

        FragU a;
        a.q[0] = Xs[cur][arow + half16 * 2];
        a.q[1] = Xs[cur][arow + half16 * 2 + 1];

        // ping-pong B fragments: loads for nsub+1 target the *other* register
        // set than the wmma for nsub reads -> no WAR chain, partial dscnt waits
        FragU bfr[2];
        {
            int r0 = l15 * ROWQ + half16 * 2;
            bfr[0].q[0] = Ws[cur][r0];
            bfr[0].q[1] = Ws[cur][r0 + 1];
        }
#pragma unroll
        for (int nsub = 0; nsub < 8; ++nsub) {
            if (nsub < 7) {
                int r = ((nsub + 1) * 16 + l15) * ROWQ + half16 * 2;
                bfr[(nsub + 1) & 1].q[0] = Ws[cur][r];
                bfr[(nsub + 1) & 1].q[1] = Ws[cur][r + 1];
            }
            acc[nsub] = __builtin_amdgcn_wmma_f32_16x16x32_bf16(
                false, a.v, false, bfr[nsub & 1].v, (short)0, acc[nsub], false, false);
        }

        if (kk + 1 < KIT) {
#if __has_builtin(__builtin_amdgcn_global_load_async_to_lds_b128)
            wait_async0();
#endif
            __syncthreads();
        }
    }

    moe_epilogue(acc, sTok, sWt, bvec, Y, e, tBase, nBase, cnt, wave, l15, half16);
}

// ---------------------------------------------------------------------------
// Kernel 3 (fallback, small ws): grouped GEMM with in-loop f32->bf16 convert
// ---------------------------------------------------------------------------
__global__ void __launch_bounds__(256, 2)
moe_gemm_f32(const float* __restrict__ X,
             const float* __restrict__ W,
             const float* __restrict__ bvec,
             const int*   __restrict__ counts,
             const int*   __restrict__ tokIds,
             const float* __restrict__ tokWt,
             float*       __restrict__ Y) {
    __shared__ uint4 Xs[2][BM * ROWQ];
    __shared__ uint4 Ws[2][BN * ROWQ];
    __shared__ int   sTok[BM];
    __shared__ float sWt[BM];
    __shared__ int   sCnt;

    const int e     = blockIdx.z;
    const int tBase = blockIdx.y * BM;
    const int nBase = blockIdx.x * BN;

    if (threadIdx.x == 0) sCnt = counts[e];
    __syncthreads();
    const int cnt = sCnt;
    if (tBase >= cnt) return;

    if (threadIdx.x < BM) {
        int r = tBase + threadIdx.x;
        if (r < cnt) {
            sTok[threadIdx.x] = tokIds[e * MOE_N + r];
            sWt [threadIdx.x] = tokWt [e * MOE_N + r];
        } else {
            sTok[threadIdx.x] = 0;
            sWt [threadIdx.x] = 0.f;
        }
    }
    __syncthreads();

    const int lane   = threadIdx.x & 31;
    const int wave   = threadIdx.x >> 5;
    const int l15    = lane & 15;
    const int half16 = lane >> 4;

    const unsigned long long wRowBase =
        ((unsigned long long)e * MOE_OUT + nBase) * MOE_IN;

    int frow[4], fkf4[4];
    const float* xptr[4];
#pragma unroll
    for (int i = 0; i < 4; ++i) {
        int c   = threadIdx.x + i * 256;
        frow[i] = c >> 3;
        fkf4[i] = c & 7;
        xptr[i] = X + (unsigned long long)sTok[frow[i]] * MOE_IN + fkf4[i] * 4;
    }

    float4 px[4], pw[4];
    auto prefetch = [&](int k0) {
#pragma unroll
        for (int i = 0; i < 4; ++i) {
            px[i] = *(const float4*)(xptr[i] + k0);
            pw[i] = *(const float4*)(W + wRowBase +
                                     (unsigned long long)frow[i] * MOE_IN + k0 + fkf4[i] * 4);
        }
    };
    auto commit = [&](int buf) {
        uint2* x2 = (uint2*)Xs[buf];
        uint2* w2 = (uint2*)Ws[buf];
#pragma unroll
        for (int i = 0; i < 4; ++i) {
            int idx = frow[i] * (2 * ROWQ) + fkf4[i];
            x2[idx] = make_uint2(pack_bf16x2(px[i].x, px[i].y),
                                 pack_bf16x2(px[i].z, px[i].w));
            w2[idx] = make_uint2(pack_bf16x2(pw[i].x, pw[i].y),
                                 pack_bf16x2(pw[i].z, pw[i].w));
        }
    };

    v8f acc[8];
    const v8f vzero = {0.f, 0.f, 0.f, 0.f, 0.f, 0.f, 0.f, 0.f};
#pragma unroll
    for (int t = 0; t < 8; ++t) acc[t] = vzero;

    const int arow = (wave * 16 + l15) * ROWQ;

    prefetch(0);
    commit(0);
    __syncthreads();

#pragma unroll 1
    for (int kk = 0; kk < KIT; ++kk) {
        const int cur = kk & 1;
        if (kk + 1 < KIT) prefetch((kk + 1) * KB);

        FragU a;
        a.q[0] = Xs[cur][arow + half16 * 2];
        a.q[1] = Xs[cur][arow + half16 * 2 + 1];

        FragU bfr[2];
        {
            int r0 = l15 * ROWQ + half16 * 2;
            bfr[0].q[0] = Ws[cur][r0];
            bfr[0].q[1] = Ws[cur][r0 + 1];
        }
#pragma unroll
        for (int nsub = 0; nsub < 8; ++nsub) {
            if (nsub < 7) {
                int r = ((nsub + 1) * 16 + l15) * ROWQ + half16 * 2;
                bfr[(nsub + 1) & 1].q[0] = Ws[cur][r];
                bfr[(nsub + 1) & 1].q[1] = Ws[cur][r + 1];
            }
            acc[nsub] = __builtin_amdgcn_wmma_f32_16x16x32_bf16(
                false, a.v, false, bfr[nsub & 1].v, (short)0, acc[nsub], false, false);
        }

        if (kk + 1 < KIT) {
            commit((kk + 1) & 1);
            __syncthreads();
        }
    }

    moe_epilogue(acc, sTok, sWt, bvec, Y, e, tBase, nBase, cnt, wave, l15, half16);
}

// ---------------------------------------------------------------------------
// Host launcher
// ---------------------------------------------------------------------------
extern "C" void kernel_launch(void* const* d_in, const int* in_sizes, int n_in,
                              void* d_out, int out_size, void* d_ws, size_t ws_size,
                              hipStream_t stream) {
    const float* X = (const float*)d_in[0];
    const float* G = (const float*)d_in[1];
    const float* W = (const float*)d_in[2];
    const float* b = (const float*)d_in[3];
    float*       Y = (float*)d_out;

    // ws layout: [1 MB routing][Xbf16 32 MB][Wbf16 64 MB]
    const size_t routeBytes = 1u << 20;
    const size_t xBytes = (size_t)MOE_N * MOE_IN * 2;
    const size_t wBytes = (size_t)MOE_E * MOE_OUT * MOE_IN * 2;

    char*  base   = (char*)d_ws;
    int*   counts = (int*)base;
    int*   tokIds = counts + 16;
    float* tokWt  = (float*)(tokIds + MOE_E * MOE_N);
    char*  Xb     = base + routeBytes;
    char*  Wb     = base + routeBytes + xBytes;

    unsigned n4 = (unsigned)((unsigned long long)MOE_N * MOE_OUT / 4);
    moe_zero<<<(n4 + 255) / 256, 256, 0, stream>>>((float4*)Y, n4, counts);
    moe_gate<<<MOE_N / 256, 256, 0, stream>>>(G, counts, tokIds, tokWt);

    dim3 grid(MOE_OUT / BN, MOE_N / BM, MOE_E);
    if (ws_size >= routeBytes + xBytes + wBytes) {
        unsigned xn4 = (unsigned)((size_t)MOE_N * MOE_IN / 4);
        unsigned wn4 = (unsigned)((size_t)MOE_E * MOE_OUT * MOE_IN / 4);
        moe_cvt<<<(xn4 + 255) / 256, 256, 0, stream>>>((const float4*)X, (uint2*)Xb, xn4);
        moe_cvt<<<(wn4 + 255) / 256, 256, 0, stream>>>((const float4*)W, (uint2*)Wb, wn4);
        moe_gemm_bf16<<<grid, 256, 0, stream>>>(Xb, Wb, b, counts, tokIds, tokWt, Y);
    } else {
        moe_gemm_f32<<<grid, 256, 0, stream>>>(X, W, b, counts, tokIds, tokWt, Y);
    }
}
